// DualFrequencyEncoder_68994354643182
// MI455X (gfx1250) — matrix operand order repaired and checked by reading, not compiled, not run on hardware
//
#include <hip/hip_runtime.h>
#include <hip/hip_bf16.h>

// Problem constants (match reference)
#define BB 2
#define LL 64
#define SS 256
#define DD 128
#define HH 4
#define MROWS (BB * LL * SS) /* 32768 */

typedef _Float16 h16;
typedef __attribute__((ext_vector_type(4)))  _Float16 v4h;
typedef __attribute__((ext_vector_type(8)))  _Float16 v8h;
typedef __attribute__((ext_vector_type(16))) _Float16 v16h;
typedef __attribute__((ext_vector_type(8)))  float    v8f;

// ---------------------------------------------------------------------------
// WMMA helpers (CDNA5 wave32, 16x16x32 f16 -> f32)
// ---------------------------------------------------------------------------
__device__ __forceinline__ v8f wmma_f16(v16h a, v16h b, v8f c) {
  // 8 args: (neg_a, A, neg_b, B, c_mod, C, reuse_a, reuse_b)
  return __builtin_amdgcn_wmma_f32_16x16x32_f16(false, a, false, b, (short)0, c,
                                                false, false);
}

__device__ __forceinline__ v16h cat8(v8h lo, v8h hi) {
  return __builtin_shufflevector(lo, hi, 0, 1, 2, 3, 4, 5, 6, 7, 8, 9, 10, 11,
                                 12, 13, 14, 15);
}

// A fragment: 16x32 f16 from row-major [m][k] storage (K contiguous).
// lanes 0-15: m=lane, K = {kb..kb+7, kb+16..kb+23} with kb = 0
// lanes 16-31: m=lane-16, kb = 8
// Two 16-byte ds loads per lane (ldm*2 and kcol0*2 must be multiples of 16B).
__device__ __forceinline__ v16h load_a_frag(const h16* base, int ldm, int mrow0,
                                            int kcol0, int lane) {
  const int m  = mrow0 + (lane & 15);
  const int kb = (lane >> 4) << 3;
  const h16* p = base + m * ldm + kcol0 + kb;
  const v8h lo = *(const v8h*)p;        // k = kb .. kb+7
  const v8h hi = *(const v8h*)(p + 16); // k = kb+16 .. kb+23
  return cat8(lo, hi);
}

// B fragment: B[k][n] sourced from K-contiguous storage base[n][k]
// (i.e. B transposed in LDS).  lanes 0-15: K={0..7,16..23}; lanes 16-31: K={8..15,24..31}.
__device__ __forceinline__ v16h load_b_fragT(const h16* base, int ldm, int ncol0,
                                             int kcol0, int lane) {
  const int n  = ncol0 + (lane & 15);
  const int kb = (lane >> 4) << 3;
  const h16* p = base + n * ldm + kcol0 + kb;
  const v8h lo = *(const v8h*)p;       // k = kb .. kb+7
  const v8h hi = *(const v8h*)(p + 16); // k = kb+16 .. kb+23
  return cat8(lo, hi);
}

__device__ __forceinline__ void zero_v8f(v8f& v) {
#pragma unroll
  for (int i = 0; i < 8; ++i) v[i] = 0.f;
}

__device__ __forceinline__ v4h cvt4(float4 f) {
  v4h h;
  h[0] = (h16)f.x; h[1] = (h16)f.y; h[2] = (h16)f.z; h[3] = (h16)f.w;
  return h;
}

// ---------------------------------------------------------------------------
// Spearman ranks: rc[b,s,l] = rank - 31.5 (ranks are a permutation of 0..63)
// ---------------------------------------------------------------------------
__global__ __launch_bounds__(64) void rank_kernel(const float* __restrict__ rets,
                                                  float* __restrict__ rc_t,
                                                  float* __restrict__ normv) {
  const int bs = blockIdx.x;  // b*256 + s
  const int b = bs >> 8, s = bs & 255;
  const int l = threadIdx.x;  // 0..63
  __shared__ float lr[64];
  __shared__ float red[64];
  lr[l] = rets[(b * LL + l) * SS + s];
  __syncthreads();
  const float x = lr[l];
  int rank = 0;
  for (int j = 0; j < LL; ++j) {
    const float xj = lr[j];
    rank += (xj < x) || (xj == x && j < l);  // stable argsort(argsort) tie rule
  }
  const float rc = (float)rank - 31.5f;
  rc_t[bs * LL + l] = rc;
  red[l] = rc * rc;
  __syncthreads();
  for (int off = 32; off > 0; off >>= 1) {
    if (l < off) red[l] += red[l + off];
    __syncthreads();
  }
  if (l == 0) normv[bs] = sqrtf(red[0]);
}

// ---------------------------------------------------------------------------
// p_spa[b,s,:] = corr_row(b,s) @ spatial_embedding
// ---------------------------------------------------------------------------
__global__ __launch_bounds__(128) void pspa_kernel(const float* __restrict__ rc_t,
                                                   const float* __restrict__ normv,
                                                   const float* __restrict__ emb,
                                                   float* __restrict__ pSpa) {
  const int bs = blockIdx.x;
  const int b = bs >> 8, s = bs & 255;
  const int t = threadIdx.x;  // 0..127
  __shared__ float rcs[64];
  __shared__ float crow[256];
  if (t < 64) rcs[t] = rc_t[bs * LL + t];
  __syncthreads();
  const float ns = normv[bs];
  for (int tt = t; tt < SS; tt += 128) {
    const float* rr = rc_t + (b * SS + tt) * LL;
    float dot = 0.f;
    for (int j = 0; j < LL; ++j) dot += rcs[j] * rr[j];
    crow[tt] = (tt == s) ? 1.0f : dot / (ns * normv[b * SS + tt]);
  }
  __syncthreads();
  float acc = 0.f;
  for (int tt = 0; tt < SS; ++tt) acc += crow[tt] * emb[tt * DD + t];
  pSpa[bs * DD + t] = acc;
}

// ---------------------------------------------------------------------------
// p_tem[b*L+l, :] = time.W[ts_index] + time.b
// ---------------------------------------------------------------------------
__global__ __launch_bounds__(128) void ptem_kernel(const int* __restrict__ ts,
                                                   const float* __restrict__ W,
                                                   const float* __restrict__ bias,
                                                   float* __restrict__ pTem) {
  const int r = blockIdx.x, d = threadIdx.x;
  const int slot = ts[r];
  pTem[r * DD + d] = W[slot * DD + d] + bias[d];
}

// ---------------------------------------------------------------------------
// Generic M x 128 x 128 GEMM via WMMA.  C = act(A @ W + bias) (+adds)
// Block: 64 rows x 128 cols, 256 threads = 8 waves; each wave 1 row-strip x 4 tiles.
// W is staged TRANSPOSED in LDS so B fragments are two ds_load_b128 per lane.
// ---------------------------------------------------------------------------
__global__ __launch_bounds__(256) void gemm128_kernel(
    const float* __restrict__ A, const float* __restrict__ W,
    const float* __restrict__ bias, float* __restrict__ C,
    const float* __restrict__ addFull, const float* __restrict__ addBL,
    const float* __restrict__ addBS, int doRelu) {
  __shared__ h16 sA[64][136];    // [m][k], 272B row stride (16B aligned)
  __shared__ h16 sWt[128][136];  // [n][k] (transposed weights)
  const int tid = threadIdx.x;
  const int row0 = blockIdx.x * 64;
  // stage A: float4 global loads, v4h LDS stores
  for (int i = tid; i < 64 * 32; i += 256) {
    const int r = i >> 5, c4 = (i & 31) << 2;
    const float4 f = *(const float4*)&A[(row0 + r) * DD + c4];
    *(v4h*)&sA[r][c4] = cvt4(f);
  }
  // stage W transposed: coalesced float4 reads, strided h16 stores
  for (int i = tid; i < 128 * 32; i += 256) {
    const int k = i >> 5, n4 = (i & 31) << 2;
    const float4 f = *(const float4*)&W[k * DD + n4];
    sWt[n4 + 0][k] = (h16)f.x;
    sWt[n4 + 1][k] = (h16)f.y;
    sWt[n4 + 2][k] = (h16)f.z;
    sWt[n4 + 3][k] = (h16)f.w;
  }
  __syncthreads();

  const int wave = tid >> 5, lane = tid & 31;
  const int strip = (wave & 3) * 16, colbase = (wave >> 2) * 64;
  v8f acc[4];
#pragma unroll
  for (int i = 0; i < 4; ++i) zero_v8f(acc[i]);

#pragma unroll
  for (int ks = 0; ks < 4; ++ks) {
    const v16h a = load_a_frag(&sA[0][0], 136, strip, ks * 32, lane);
#pragma unroll
    for (int ct = 0; ct < 4; ++ct) {
      const v16h b =
          load_b_fragT(&sWt[0][0], 136, colbase + ct * 16, ks * 32, lane);
      acc[ct] = wmma_f16(a, b, acc[ct]);
    }
  }

#pragma unroll
  for (int ct = 0; ct < 4; ++ct) {
    const int col = colbase + ct * 16 + (lane & 15);
#pragma unroll
    for (int r = 0; r < 8; ++r) {
      const int row = row0 + strip + r + ((lane >> 4) << 3);
      float v = acc[ct][r] + bias[col];
      if (doRelu) v = fmaxf(v, 0.f);
      if (addFull) v += addFull[row * DD + col];
      if (addBL || addBS) {
        const int bb = row >> 14, l = (row >> 8) & 63, s = row & 255;
        if (addBL) v += addBL[(bb * LL + l) * DD + col];
        if (addBS) v += addBS[(bb * SS + s) * DD + col];
      }
      C[row * DD + col] = v;
    }
  }
}

// ---------------------------------------------------------------------------
// Dilated conv (taps at l-1, l+1) as two shifted WMMA GEMMs.
// out = relu(conv + b) + p_tem + p_spa
// ---------------------------------------------------------------------------
__global__ __launch_bounds__(256) void conv2_kernel(
    const float* __restrict__ Xh, const float* __restrict__ wconv,
    const float* __restrict__ bias, const float* __restrict__ pTem,
    const float* __restrict__ pSpa, float* __restrict__ C) {
  __shared__ h16 sA[64][136];
  __shared__ h16 sWt[128][136];  // [out][in] (== B transposed: B[k=in][n=out])
  const int tid = threadIdx.x;
  const int row0 = blockIdx.x * 64;
  const int wave = tid >> 5, lane = tid & 31;
  const int strip = (wave & 3) * 16, colbase = (wave >> 2) * 64;
  v8f acc[4];
#pragma unroll
  for (int i = 0; i < 4; ++i) zero_v8f(acc[i]);

  for (int tap = 0; tap < 2; ++tap) {
    const int shift = (tap == 0) ? -SS : SS;
    for (int i = tid; i < 64 * 32; i += 256) {
      const int r = i >> 5, c4 = (i & 31) << 2;
      const int row = row0 + r;
      const int l = (row >> 8) & 63;
      const bool valid = (tap == 0) ? (l > 0) : (l < LL - 1);
      v4h h;
      if (valid) {
        const float4 f = *(const float4*)&Xh[(row + shift) * DD + c4];
        h = cvt4(f);
      } else {
        h[0] = h[1] = h[2] = h[3] = (h16)0.f;
      }
      *(v4h*)&sA[r][c4] = h;
    }
    // w is [O][I][2]; float4 covers taps {ii,ii+1} for one output channel
    for (int i = tid; i < 128 * 64; i += 256) {
      const int oo = i >> 6, ii2 = (i & 63) << 1;
      const float4 f = *(const float4*)&wconv[(oo * DD + ii2) * 2];
      sWt[oo][ii2 + 0] = (h16)(tap == 0 ? f.x : f.y);
      sWt[oo][ii2 + 1] = (h16)(tap == 0 ? f.z : f.w);
    }
    __syncthreads();
#pragma unroll
    for (int ks = 0; ks < 4; ++ks) {
      const v16h a = load_a_frag(&sA[0][0], 136, strip, ks * 32, lane);
#pragma unroll
      for (int ct = 0; ct < 4; ++ct) {
        const v16h b =
            load_b_fragT(&sWt[0][0], 136, colbase + ct * 16, ks * 32, lane);
        acc[ct] = wmma_f16(a, b, acc[ct]);
      }
    }
    __syncthreads();
  }

#pragma unroll
  for (int ct = 0; ct < 4; ++ct) {
    const int col = colbase + ct * 16 + (lane & 15);
#pragma unroll
    for (int r = 0; r < 8; ++r) {
      const int row = row0 + strip + r + ((lane >> 4) << 3);
      const int bb = row >> 14, l = (row >> 8) & 63, s = row & 255;
      float v = fmaxf(acc[ct][r] + bias[col], 0.f);
      v += pTem[(bb * LL + l) * DD + col] + pSpa[(bb * SS + s) * DD + col];
      C[row * DD + col] = v;
    }
  }
}

// ---------------------------------------------------------------------------
// Fused attention: one block per (n, h, 16-row strip).  T = 64 or 256.
// Q/K/V are [n*T + t, 128] with head slice h*32..+32.  Output O same layout.
// V is staged transposed so attn@V B fragments are vector ds loads.
// ---------------------------------------------------------------------------
__global__ __launch_bounds__(128) void attn_kernel(const float* __restrict__ Q,
                                                   const float* __restrict__ K,
                                                   const float* __restrict__ V,
                                                   float* __restrict__ O, int T) {
  __shared__ h16 lq[16][32];     // [row][d]
  __shared__ h16 lk[256][32];    // [t][d]  (== B^T layout for QK^T)
  __shared__ h16 lvT[32][264];   // [d][t]  (== B^T layout for attn@V), 528B rows
  __shared__ float sc[16][256];  // scores, later packed f16 probabilities
  __shared__ float lo[16][32];
  const int tid = threadIdx.x, wave = tid >> 5, lane = tid & 31;
  const int strips = T >> 4;
  const int id = blockIdx.x;
  const int strip = id % strips;
  const int nh = id / strips;
  const int h = nh & 3, n = nh >> 2;
  const float scale = 0.17677669529663687f;  // 1/sqrt(32)

  const int qr0 = (n * T + strip * 16) * DD + h * 32;
  for (int i = tid; i < 16 * 8; i += 128) {
    const int r = i >> 3, d4 = (i & 7) << 2;
    const float4 f = *(const float4*)&Q[qr0 + r * DD + d4];
    v4h hq;
    hq[0] = (h16)(f.x * scale); hq[1] = (h16)(f.y * scale);
    hq[2] = (h16)(f.z * scale); hq[3] = (h16)(f.w * scale);
    *(v4h*)&lq[r][d4] = hq;
  }
  const int kv0 = n * T * DD + h * 32;
  for (int i = tid; i < T * 8; i += 128) {
    const int t = i >> 3, d4 = (i & 7) << 2;
    const float4 fk = *(const float4*)&K[kv0 + t * DD + d4];
    *(v4h*)&lk[t][d4] = cvt4(fk);
    const float4 fv = *(const float4*)&V[kv0 + t * DD + d4];
    lvT[d4 + 0][t] = (h16)fv.x;
    lvT[d4 + 1][t] = (h16)fv.y;
    lvT[d4 + 2][t] = (h16)fv.z;
    lvT[d4 + 3][t] = (h16)fv.w;
  }
  for (int i = tid; i < 512; i += 128) lo[i >> 5][i & 31] = 0.f;
  __syncthreads();

  // scores: strip(16) x T, K = head_dim = 32 -> one WMMA per 16x16 tile
  {
    const v16h a = load_a_frag(&lq[0][0], 32, 0, 0, lane);
    for (int ct = wave; ct < strips; ct += 4) {  // wave-uniform loop, EXEC full
      v8f c;
      zero_v8f(c);
      const v16h b = load_b_fragT(&lk[0][0], 32, ct * 16, 0, lane);
      c = wmma_f16(a, b, c);
      const int col = ct * 16 + (lane & 15);
#pragma unroll
      for (int r = 0; r < 8; ++r) sc[r + ((lane >> 4) << 3)][col] = c[r];
    }
  }
  __syncthreads();

  // softmax per row; repack probabilities in place as f16 (same-thread ordering safe)
  if (tid < 16) {
    float mx = -3.4e38f;
    for (int c = 0; c < T; ++c) mx = fmaxf(mx, sc[tid][c]);
    float sum = 0.f;
    for (int c = 0; c < T; ++c) {
      const float e = __expf(sc[tid][c] - mx);
      sc[tid][c] = e;
      sum += e;
    }
    const float inv = 1.f / sum;
    h16* hp = (h16*)&sc[tid][0];
    for (int c = 0; c < T; ++c) hp[c] = (h16)(sc[tid][c] * inv);
  }
  __syncthreads();

  // o = attn @ v ; K dimension (= T) split across waves, reduced via ds_add_f32
  {
    const int ksteps = T >> 5;
    v8f accp[2];
    zero_v8f(accp[0]);
    zero_v8f(accp[1]);
    for (int ks = wave; ks < ksteps; ks += 4) {  // wave-uniform
      const v16h a = load_a_frag((const h16*)&sc[0][0], 512, 0, ks * 32, lane);
#pragma unroll
      for (int nt = 0; nt < 2; ++nt) {
        const v16h b =
            load_b_fragT(&lvT[0][0], 264, nt * 16, ks * 32, lane);
        accp[nt] = wmma_f16(a, b, accp[nt]);
      }
    }
#pragma unroll
    for (int nt = 0; nt < 2; ++nt) {
      const int col = nt * 16 + (lane & 15);
#pragma unroll
      for (int r = 0; r < 8; ++r)
        atomicAdd(&lo[r + ((lane >> 4) << 3)][col], accp[nt][r]);
    }
  }
  __syncthreads();

  for (int i = tid; i < 512; i += 128) {
    const int r = i >> 5, d = i & 31;
    O[qr0 + r * DD + d] = lo[r][d];
  }
}

// ---------------------------------------------------------------------------
// LayerNorm over D=128 per row
// ---------------------------------------------------------------------------
__global__ __launch_bounds__(128) void ln_kernel(const float* __restrict__ x,
                                                 const float* __restrict__ g,
                                                 const float* __restrict__ b,
                                                 float* __restrict__ y) {
  const int row = blockIdx.x, t = threadIdx.x;
  __shared__ float red[128];
  const float v = x[row * DD + t];
  red[t] = v;
  __syncthreads();
  for (int off = 64; off > 0; off >>= 1) {
    if (t < off) red[t] += red[t + off];
    __syncthreads();
  }
  const float mu = red[0] * (1.f / 128.f);
  __syncthreads();
  const float d = v - mu;
  red[t] = d * d;
  __syncthreads();
  for (int off = 64; off > 0; off >>= 1) {
    if (t < off) red[t] += red[t + off];
    __syncthreads();
  }
  const float var = red[0] * (1.f / 128.f);
  y[row * DD + t] = d * rsqrtf(var + 1e-5f) * g[t] + b[t];
}

// ---------------------------------------------------------------------------
// Host orchestration
// ---------------------------------------------------------------------------
extern "C" void kernel_launch(void* const* d_in, const int* in_sizes, int n_in,
                              void* d_out, int out_size, void* d_ws, size_t ws_size,
                              hipStream_t stream) {
  (void)in_sizes; (void)n_in; (void)out_size; (void)ws_size;
  const float* Xl   = (const float*)d_in[0];
  const float* Xh   = (const float*)d_in[1];
  const int*   ts   = (const int*)d_in[2];
  const float* rets = (const float*)d_in[3];
  // temporal MHA params (dict insertion order)
  const float* tWq = (const float*)d_in[4],  *tbq = (const float*)d_in[5];
  const float* tWk = (const float*)d_in[6],  *tbk = (const float*)d_in[7];
  const float* tWv = (const float*)d_in[8],  *tbv = (const float*)d_in[9];
  const float* tWo = (const float*)d_in[10], *tbo = (const float*)d_in[11];
  const float* convw = (const float*)d_in[12], *convb = (const float*)d_in[13];
  const float* timeW = (const float*)d_in[14], *timeb = (const float*)d_in[15];
  const float* emb   = (const float*)d_in[16];

  struct GatP {
    const float *Wq, *bq, *Wk, *bk, *Wv, *bv, *Wo, *bo, *lng, *lnb, *W1, *b1, *W2, *b2;
  };
  auto getGat = [&](int base) {
    GatP g;
    g.Wq = (const float*)d_in[base + 0];  g.bq = (const float*)d_in[base + 1];
    g.Wk = (const float*)d_in[base + 2];  g.bk = (const float*)d_in[base + 3];
    g.Wv = (const float*)d_in[base + 4];  g.bv = (const float*)d_in[base + 5];
    g.Wo = (const float*)d_in[base + 6];  g.bo = (const float*)d_in[base + 7];
    g.lng = (const float*)d_in[base + 8]; g.lnb = (const float*)d_in[base + 9];
    g.W1 = (const float*)d_in[base + 10]; g.b1 = (const float*)d_in[base + 11];
    g.W2 = (const float*)d_in[base + 12]; g.b2 = (const float*)d_in[base + 13];
    return g;
  };
  const GatP gl = getGat(17);
  const GatP gh = getGat(31);

  // workspace layout (floats): 5 big [32768,128] buffers + small buffers
  float* wsf = (float*)d_ws;
  const size_t NB = (size_t)MROWS * DD;  // 4,194,304
  float* buf0 = wsf;
  float* buf1 = wsf + NB;
  float* buf2 = wsf + 2 * NB;
  float* buf3 = wsf + 3 * NB;
  float* buf4 = wsf + 4 * NB;
  float* rc   = wsf + 5 * NB;                  // [B*S, L]
  float* nv   = rc + (size_t)BB * SS * LL;
  float* pSpa = nv + (size_t)BB * SS;          // [B*S, D]
  float* pTem = pSpa + (size_t)BB * SS * DD;   // [B*L, D]
  float* outF = (float*)d_out;

  auto gemm = [&](const float* A, const float* W, const float* bias, float* C,
                  const float* addFull, const float* addBL, const float* addBS,
                  int relu) {
    gemm128_kernel<<<dim3(MROWS / 64), dim3(256), 0, stream>>>(
        A, W, bias, C, addFull, addBL, addBS, relu);
  };

  // --- graph embeddings ---
  rank_kernel<<<dim3(BB * SS), dim3(64), 0, stream>>>(rets, rc, nv);
  ptem_kernel<<<dim3(BB * LL), dim3(128), 0, stream>>>(ts, timeW, timeb, pTem);
  pspa_kernel<<<dim3(BB * SS), dim3(128), 0, stream>>>(rc, nv, emb, pSpa);

  // --- temporal self-attention branch (N = B*S = 512, T = L = 64) ---
  gemm(Xl, tWq, tbq, buf0, nullptr, nullptr, nullptr, 0);
  gemm(Xl, tWk, tbk, buf1, nullptr, nullptr, nullptr, 0);
  gemm(Xl, tWv, tbv, buf2, nullptr, nullptr, nullptr, 0);
  attn_kernel<<<dim3(BB * SS * HH * (LL / 16)), dim3(128), 0, stream>>>(
      buf0, buf1, buf2, buf3, LL);
  // Xl_in = O @ Wo + bo + p_tem + p_spa  -> buf4
  gemm(buf3, tWo, tbo, buf4, nullptr, pTem, pSpa, 0);

  auto runGat = [&](const float* Xin, const GatP& g, float* out) {
    gemm(Xin, g.Wq, g.bq, buf0, nullptr, nullptr, nullptr, 0);
    gemm(Xin, g.Wk, g.bk, buf1, nullptr, nullptr, nullptr, 0);
    gemm(Xin, g.Wv, g.bv, buf2, nullptr, nullptr, nullptr, 0);
    attn_kernel<<<dim3(BB * LL * HH * (SS / 16)), dim3(128), 0, stream>>>(
        buf0, buf1, buf2, buf3, SS);
    gemm(buf3, g.Wo, g.bo, buf0, Xin, nullptr, nullptr, 0);  // Yin = out + x
    ln_kernel<<<dim3(MROWS), dim3(128), 0, stream>>>(buf0, g.lng, g.lnb, buf1);
    gemm(buf1, g.W1, g.b1, buf2, nullptr, nullptr, nullptr, 1);  // relu(y@W1+b1)
    gemm(buf2, g.W2, g.b2, out, buf1, nullptr, nullptr, 0);      // ff + y
  };

  // --- GAT on low-frequency branch -> first output ---
  runGat(buf4, gl, outF);

  // --- dilated conv branch -> Xh_in (buf4 free after GAT-l consumed it) ---
  conv2_kernel<<<dim3(MROWS / 64), dim3(256), 0, stream>>>(Xh, convw, convb,
                                                           pTem, pSpa, buf4);

  // --- GAT on high-frequency branch -> second output ---
  runGat(buf4, gh, outF + NB);
}
